// Joint_962072674642
// MI455X (gfx1250) — matrix operand order repaired and checked by reading, not compiled, not run on hardware
//
#include <hip/hip_runtime.h>
#include <hip/hip_bf16.h>
#include <math.h>

typedef __attribute__((ext_vector_type(16))) _Float16 v16h;
typedef __attribute__((ext_vector_type(8)))  float    v8f;

#define B_   8
#define T_   256
#define U_   64
#define H_   512
#define HID_ 640
#define V_   1024
#define KS1  16   // H/32
#define KS2  20   // HID/32

#if defined(__has_builtin)
#  if __has_builtin(__builtin_amdgcn_tanhf)
#    define TANHF(x) __builtin_amdgcn_tanhf(x)
#  endif
#endif
#ifndef TANHF
#  define TANHF(x) tanhf(x)
#endif

// ---------------------------------------------------------------------------
// Pack a fp32 weight matrix W[n, k] (row stride rowStride, starting at colOff)
// into f16 fragment-major layout for v_wmma_f32_16x16x32_f16 B operand:
//   packed[((ntile*KS + ks)*32 + lane)*16 + s] = W[ntile*16 + lane%16,
//                                   ks*32 + 8*(lane/16) + s + 8*(s/8)]
// (B layout: lanes 0-15 hold K=0-7,16-23, lanes 16-31 hold K=8-15,24-31.)
// ---------------------------------------------------------------------------
__global__ void pack_w_kernel(const float* __restrict__ W, int rowStride,
                              int colOff, int KS, _Float16* __restrict__ out,
                              int total) {
  int id = blockIdx.x * 256 + threadIdx.x;
  if (id >= total) return;
  int s     = id & 15;
  int lane  = (id >> 4) & 31;
  int ks    = (id >> 9) % KS;
  int ntile = id / (KS << 9);
  int k = ks * 32 + ((lane >> 4) << 3) + s + ((s >> 3) << 3);
  int n = (ntile << 4) + (lane & 15);
  out[id] = (_Float16)W[(size_t)n * rowStride + colOff + k];
}

// ---------------------------------------------------------------------------
// Projection GEMM: P[row, n] = sum_k A[row, k] * W[n, k]   (no bias)
// A: fp32 [rows, K] row-major (K is also the row stride).
// Wp: packed f16 fragments (layout above). P: fp32 [rows, N].
// One wave per 16x16 output tile; block = 8 waves = 8 consecutive n-tiles.
// ---------------------------------------------------------------------------
__global__ void __launch_bounds__(256) proj_kernel(
    const float* __restrict__ A, const _Float16* __restrict__ Wp,
    float* __restrict__ P, int K, int N, int nTileBlk) {
  const int wave = threadIdx.x >> 5;
  const int lane = threadIdx.x & 31;
  const int half = lane >> 4;
  const int m    = lane & 15;

  const int rowTile = blockIdx.x / nTileBlk;
  const int nTile   = (blockIdx.x % nTileBlk) * 8 + wave;
  const int row0    = rowTile * 16;
  const int KS      = K >> 5;

  const float*    arow  = A + (size_t)(row0 + m) * K;
  const _Float16* wbase = Wp + (size_t)nTile * KS * 512 + lane * 16;

  v8f acc = {};
  for (int ks = 0; ks < KS; ++ks) {
    const int kb = ks * 32 + half * 8;
    float4 c0 = *(const float4*)(arow + kb);
    float4 c1 = *(const float4*)(arow + kb + 4);
    float4 c2 = *(const float4*)(arow + kb + 16);
    float4 c3 = *(const float4*)(arow + kb + 20);
    v16h a;
    a[0]  = (_Float16)c0.x; a[1]  = (_Float16)c0.y;
    a[2]  = (_Float16)c0.z; a[3]  = (_Float16)c0.w;
    a[4]  = (_Float16)c1.x; a[5]  = (_Float16)c1.y;
    a[6]  = (_Float16)c1.z; a[7]  = (_Float16)c1.w;
    a[8]  = (_Float16)c2.x; a[9]  = (_Float16)c2.y;
    a[10] = (_Float16)c2.z; a[11] = (_Float16)c2.w;
    a[12] = (_Float16)c3.x; a[13] = (_Float16)c3.y;
    a[14] = (_Float16)c3.z; a[15] = (_Float16)c3.w;
    v16h b = *(const v16h*)(wbase + (size_t)ks * 512);
    acc = __builtin_amdgcn_wmma_f32_16x16x32_f16(false, a, false, b,
                                                 (short)0, acc, false, false);
  }

  float* prow = P + (size_t)(row0 + half * 8) * N + nTile * 16 + m;
#pragma unroll
  for (int r = 0; r < 8; ++r) prow[(size_t)r * N] = acc[r];
}

// ---------------------------------------------------------------------------
// Fused joint kernel. Block = (b, t, u-tile of 16). 256 threads = 8 waves.
// Phase 1: compute tanh(enc_p[t,k] + dec_p[u,k] + b1[k]) once for the whole
//          16x640 A tile, written into LDS directly in WMMA A-fragment layout.
// Phase 2: each wave owns 128 columns of V (8 n-tiles), streams packed W2
//          fragments and issues 16x16x32 f16 WMMAs with fp32 accumulation.
// ---------------------------------------------------------------------------
__global__ void __launch_bounds__(256) joint_kernel(
    const float* __restrict__ encP, const float* __restrict__ decP,
    const float* __restrict__ b1, const _Float16* __restrict__ W2p,
    const float* __restrict__ b2, float* __restrict__ out) {
  __shared__ _Float16 afrag[KS2 * 512];  // 20 KB

  const int tid  = threadIdx.x;
  const int wave = tid >> 5;
  const int lane = tid & 31;
  const int half = lane >> 4;
  const int m    = lane & 15;

  const int blk  = blockIdx.x;
  const int b    = blk >> 10;        // T_*(U_/16) = 1024 tiles per batch
  const int tile = blk & 1023;
  const int t    = tile >> 2;
  const int u0   = (tile & 3) << 4;

  const float* erow = encP + (size_t)(b * T_ + t) * HID_;
  const float* drow = decP + (size_t)(b * U_ + u0 + m) * HID_;

  // ---- Phase 1: build A fragments (each tanh computed exactly once) ----
  for (int ks = wave; ks < KS2; ks += 8) {
    const int kb = ks * 32 + half * 8;
    float4 e0 = *(const float4*)(erow + kb);
    float4 e1 = *(const float4*)(erow + kb + 4);
    float4 e2 = *(const float4*)(erow + kb + 16);
    float4 e3 = *(const float4*)(erow + kb + 20);
    float4 d0 = *(const float4*)(drow + kb);
    float4 d1 = *(const float4*)(drow + kb + 4);
    float4 d2 = *(const float4*)(drow + kb + 16);
    float4 d3 = *(const float4*)(drow + kb + 20);
    float4 g0 = *(const float4*)(b1 + kb);
    float4 g1 = *(const float4*)(b1 + kb + 4);
    float4 g2 = *(const float4*)(b1 + kb + 16);
    float4 g3 = *(const float4*)(b1 + kb + 20);
    v16h a;
    a[0]  = (_Float16)TANHF(e0.x + d0.x + g0.x);
    a[1]  = (_Float16)TANHF(e0.y + d0.y + g0.y);
    a[2]  = (_Float16)TANHF(e0.z + d0.z + g0.z);
    a[3]  = (_Float16)TANHF(e0.w + d0.w + g0.w);
    a[4]  = (_Float16)TANHF(e1.x + d1.x + g1.x);
    a[5]  = (_Float16)TANHF(e1.y + d1.y + g1.y);
    a[6]  = (_Float16)TANHF(e1.z + d1.z + g1.z);
    a[7]  = (_Float16)TANHF(e1.w + d1.w + g1.w);
    a[8]  = (_Float16)TANHF(e2.x + d2.x + g2.x);
    a[9]  = (_Float16)TANHF(e2.y + d2.y + g2.y);
    a[10] = (_Float16)TANHF(e2.z + d2.z + g2.z);
    a[11] = (_Float16)TANHF(e2.w + d2.w + g2.w);
    a[12] = (_Float16)TANHF(e3.x + d3.x + g3.x);
    a[13] = (_Float16)TANHF(e3.y + d3.y + g3.y);
    a[14] = (_Float16)TANHF(e3.z + d3.z + g3.z);
    a[15] = (_Float16)TANHF(e3.w + d3.w + g3.w);
    *(v16h*)(afrag + ((ks << 5) + lane) * 16) = a;
  }
  __syncthreads();

  // ---- Phase 2: 16 x 128 GEMM tile per wave over K = 640 ----
  v8f acc[8];
#pragma unroll
  for (int nt = 0; nt < 8; ++nt) acc[nt] = (v8f){};

  const _Float16* wbase = W2p + (size_t)(wave * 8) * KS2 * 512 + lane * 16;
  for (int ks = 0; ks < KS2; ++ks) {
    v16h a = *(const v16h*)(afrag + ((ks << 5) + lane) * 16);
#pragma unroll
    for (int nt = 0; nt < 8; ++nt) {
      v16h bf = *(const v16h*)(wbase + (size_t)(nt * KS2 + ks) * 512);
      acc[nt] = __builtin_amdgcn_wmma_f32_16x16x32_f16(false, a, false, bf,
                                                       (short)0, acc[nt],
                                                       false, false);
    }
  }

  // ---- Epilogue: add b2, store fp32 output ----
  const size_t outRow0 =
      (((size_t)(b * T_ + t)) * U_ + u0 + half * 8) * V_;
  const int n00 = wave * 128 + m;
#pragma unroll
  for (int nt = 0; nt < 8; ++nt) {
    const int n = n00 + nt * 16;
    const float bias = b2[n];
#pragma unroll
    for (int r = 0; r < 8; ++r)
      out[outRow0 + (size_t)r * V_ + n] = acc[nt][r] + bias;
  }
}

// ---------------------------------------------------------------------------
extern "C" void kernel_launch(void* const* d_in, const int* in_sizes, int n_in,
                              void* d_out, int out_size, void* d_ws,
                              size_t ws_size, hipStream_t stream) {
  const float* h_enc = (const float*)d_in[0];  // [B,T,H]
  const float* h_dec = (const float*)d_in[1];  // [B,U,H]
  const float* W1    = (const float*)d_in[2];  // [HID, 2H]
  const float* b1    = (const float*)d_in[3];  // [HID]
  const float* W2    = (const float*)d_in[4];  // [V, HID]
  const float* b2    = (const float*)d_in[5];  // [V]
  float* out = (float*)d_out;

  // Workspace carve-up (~8.75 MB total)
  float*    encP = (float*)d_ws;                   // B*T*HID  = 1310720 f32
  float*    decP = encP + (size_t)B_ * T_ * HID_;  // B*U*HID  =  327680 f32
  _Float16* w1e  = (_Float16*)(decP + (size_t)B_ * U_ * HID_);  // 327680 f16
  _Float16* w1d  = w1e + (size_t)HID_ * H_;                     // 327680 f16
  _Float16* w2p  = w1d + (size_t)HID_ * H_;                     // 655360 f16

  // Pack weights into WMMA B-fragment layout (f16)
  const int tot1 = HID_ * H_;   // 327680
  const int tot2 = V_ * HID_;   // 655360
  pack_w_kernel<<<(tot1 + 255) / 256, 256, 0, stream>>>(W1, 2 * H_, 0,   KS1, w1e, tot1);
  pack_w_kernel<<<(tot1 + 255) / 256, 256, 0, stream>>>(W1, 2 * H_, H_,  KS1, w1d, tot1);
  pack_w_kernel<<<(tot2 + 255) / 256, 256, 0, stream>>>(W2, HID_,   0,   KS2, w2p, tot2);

  // Projections: enc_p [B*T, HID], dec_p [B*U, HID]
  proj_kernel<<<(B_ * T_ / 16) * 5, 256, 0, stream>>>(h_enc, w1e, encP, H_, HID_, 5);
  proj_kernel<<<(B_ * U_ / 16) * 5, 256, 0, stream>>>(h_dec, w1d, decP, H_, HID_, 5);

  // Fused tanh + joint GEMM: one block per (b, t, u-tile)
  joint_kernel<<<B_ * T_ * (U_ / 16), 256, 0, stream>>>(encP, decP, b1, w2p,
                                                        b2, out);
}
